// iTransformerMHSA_60782377173531
// MI455X (gfx1250) — matrix operand order repaired and checked by reading, not compiled, not run on hardware
//
#include <hip/hip_runtime.h>
#include <hip/hip_bf16.h>

// ---------------------------------------------------------------------------
// iTransformerMHSA for MI455X (gfx1250), bf16 WMMA pipeline.
// A=1024 assets (attention axis), B=16, D=512, 8 heads x 64.
// Roofline: ~69 GFLOP vs ~150 MB HBM traffic -> compute-bound on WMMA;
// all GEMMs via v_wmma_f32_16x16x32_bf16 (fp32 accum), flash-style attention
// so the 512MB logits tensor never exists. 2-deep software pipeline with
// parity-rotated buffers; sched_barrier pins prefetches a full step early.
// ---------------------------------------------------------------------------

#define A_DIM 1024
#define B_DIM 16
#define DMODEL 512
#define NHEAD 8
#define HDIM 64
#define MROWS (A_DIM * B_DIM)   // 16384

#if __has_builtin(__builtin_amdgcn_sched_barrier)
#define SCHED_BARRIER() __builtin_amdgcn_sched_barrier(0)
#else
#define SCHED_BARRIER()
#endif

typedef __attribute__((ext_vector_type(16))) __bf16 v16bf;
typedef __attribute__((ext_vector_type(8)))  float  v8f;

union FragBF {
    v16bf v;
    unsigned int u[8];
};

// A-matrix 16x32 bf16 fragment (row-major source [m][k], ld elements).
// lane L -> m = L%16, half = L/16;
// VGPR p (p<4): k = 2p + 8*half ; (p>=4): k = 16 + 2(p-4) + 8*half.
// Per-lane bytes form two contiguous 16B runs -> 2x global_load_b128.
__device__ inline v16bf load_afrag(const __bf16* base, int ld, int lane) {
    FragBF f;
    const int m = lane & 15, half = lane >> 4;
    const __bf16* row = base + m * ld;
#pragma unroll
    for (int p = 0; p < 8; ++p) {
        const int k = (p < 4) ? (2 * p + 8 * half) : (16 + 2 * (p - 4) + 8 * half);
        f.u[p] = *(const unsigned int*)(row + k);
    }
    return f.v;
}

// B-matrix 32x16 bf16 fragment; operand stored transposed as Bt[n][k].
// lane L -> n = L%16; VGPR p (p<4): k = 2p + 8*half ; (p>=4): k = 16 + 2(p-4) + 8*half.
__device__ inline v16bf load_bfrag(const __bf16* baseT, int ld, int lane) {
    FragBF f;
    const int n = lane & 15, half = lane >> 4;
    const __bf16* row = baseT + n * ld;
#pragma unroll
    for (int p = 0; p < 8; ++p) {
        const int k = (p < 4) ? (2 * p + 8 * half) : (16 + 2 * (p - 4) + 8 * half);
        f.u[p] = *(const unsigned int*)(row + k);
    }
    return f.v;
}

__device__ inline v8f wmma_bf16(v16bf a, v16bf b, v8f c) {
    return __builtin_amdgcn_wmma_f32_16x16x32_bf16(
        /*neg_a=*/false, a, /*neg_b=*/false, b,
        /*c_mod=*/(short)0, c, /*reuse_a=*/false, /*reuse_b=*/false);
}

// ---------------------------------------------------------------------------
// fp32 -> bf16 conversion, 4 elements per thread, vectorized accesses.
// ---------------------------------------------------------------------------
__global__ void cvt_kernel(const float* __restrict__ in, __bf16* __restrict__ out, int n) {
    const int i4 = blockIdx.x * blockDim.x + threadIdx.x;
    if (i4 * 4 + 3 < n) {
        const float4 f = ((const float4*)in)[i4];
        union { __bf16 h[4]; uint2 u; } r;
        r.h[0] = (__bf16)f.x; r.h[1] = (__bf16)f.y;
        r.h[2] = (__bf16)f.z; r.h[3] = (__bf16)f.w;
        ((uint2*)out)[i4] = r.u;
    }
}

// ---------------------------------------------------------------------------
// GEMM: Y[m][n] = sum_k X[m][k] * W[n][k] + bias[n]
// M = 16384, N = 512, K = 512.  Block = 256 threads = 8 waves; all 8 waves
// of a block share one 16-row X tile (L0-friendly) and cover all 512 N cols.
// Fully unrolled K loop, 2-deep pipeline: step s consumes buffer (s&1) and
// reloads it for step s+2; sched_barrier keeps loads a full step early.
// mode 0/1: q/k layout [b][h][a][d] bf16 ; mode 2: v^T [b][h][d][a] bf16 ;
// mode 3: fp32 flat [m][n] (final output projection).
// ---------------------------------------------------------------------------
__global__ void proj_kernel(const __bf16* __restrict__ X, const __bf16* __restrict__ W,
                            const float* __restrict__ bias,
                            __bf16* __restrict__ outb, float* __restrict__ outf, int mode) {
    const int lane = threadIdx.x & 31;
    const int wave = threadIdx.x >> 5;
    const int g    = blockIdx.x * 8 + wave;   // 8192 waves total
    const int mtile = g >> 3;                 // 0..1023
    const int ngrp  = g & 7;                  // 0..7
    const int mbase = mtile * 16;
    const int nbase = ngrp * 64;

    v8f acc[4] = {};

    v16bf a_buf[2];
    v16bf b_buf[2][4];
#pragma unroll
    for (int s = 0; s < 2; ++s) {
        a_buf[s] = load_afrag(X + mbase * DMODEL + s * 32, DMODEL, lane);
#pragma unroll
        for (int t = 0; t < 4; ++t)
            b_buf[s][t] = load_bfrag(W + (nbase + t * 16) * DMODEL + s * 32, DMODEL, lane);
    }

#pragma unroll
    for (int s = 0; s < DMODEL / 32; ++s) {
        const int cur = s & 1;
#pragma unroll
        for (int t = 0; t < 4; ++t)
            acc[t] = wmma_bf16(a_buf[cur], b_buf[cur][t], acc[t]);
        const int kpre = (s + 2) * 32;
        if (kpre < DMODEL) {
            // WAR on buf[cur] keeps these below the WMMAs above; the barrier
            // keeps them from sinking into the next step.
            a_buf[cur] = load_afrag(X + mbase * DMODEL + kpre, DMODEL, lane);
#pragma unroll
            for (int t = 0; t < 4; ++t)
                b_buf[cur][t] = load_bfrag(W + (nbase + t * 16) * DMODEL + kpre, DMODEL, lane);
        }
        SCHED_BARRIER();
    }

    const int nl = lane & 15, half = lane >> 4;
#pragma unroll
    for (int t = 0; t < 4; ++t) {
#pragma unroll
        for (int r = 0; r < 8; ++r) {
            const int m = mbase + r + 8 * half;      // 0..16383 (= a*16 + b)
            const int n = nbase + t * 16 + nl;       // 0..511
            const float y = acc[t][r] + bias[n];
            if (mode == 3) {
                outf[m * DMODEL + n] = y;
            } else {
                const int a_ = m >> 4, b_ = m & 15;
                const int h = n >> 6, d = n & 63;
                if (mode == 2)  // V transposed: [b][h][d][a]
                    outb[((b_ * NHEAD + h) * HDIM + d) * A_DIM + a_] = (__bf16)y;
                else            // Q/K: [b][h][a][d]
                    outb[((b_ * NHEAD + h) * A_DIM + a_) * HDIM + d] = (__bf16)y;
            }
        }
    }
}

// ---------------------------------------------------------------------------
// Flash attention over the asset axis.
// grid = 128 (b,h) * 16 q-blocks; block = 128 threads = 4 waves.
// Each wave owns 16 query rows; key loop in blocks of 32, unrolled x2 with
// parity-owned K and V fragment buffers (2-deep pipeline, no copies);
// sched_barrier pins the prefetches so they overlap softmax / LDS work.
// out: bf16 [a][b][h*64+d]  (row-major [16384][512]) for the final projection.
// ---------------------------------------------------------------------------
__global__ void attn_kernel(const __bf16* __restrict__ Q, const __bf16* __restrict__ K,
                            const __bf16* __restrict__ VT, const float* __restrict__ bias,
                            __bf16* __restrict__ out) {
    __shared__ __bf16 lds_p[4][16 * 32];

    const int lane = threadIdx.x & 31;
    const int wave = threadIdx.x >> 5;
    const int bh   = blockIdx.x >> 4;            // 0..127 = b*8 + h
    const int qblk = blockIdx.x & 15;            // 0..15
    const int qbase = qblk * 64 + wave * 16;     // query row base for this wave
    const int b_ = bh >> 3, h = bh & 7;
    const int nl = lane & 15, half = lane >> 4;

    const __bf16* q  = Q  + bh * A_DIM * HDIM;
    const __bf16* k  = K  + bh * A_DIM * HDIM;
    const __bf16* vt = VT + bh * HDIM * A_DIM;

    // Q tile 16x64 as two A fragments (hoisted out of the key loop).
    const v16bf qa0 = load_afrag(q + qbase * HDIM + 0,  HDIM, lane);
    const v16bf qa1 = load_afrag(q + qbase * HDIM + 32, HDIM, lane);

    float run_m[8], run_l[8];
#pragma unroll
    for (int r = 0; r < 8; ++r) { run_m[r] = -1e30f; run_l[r] = 0.0f; }
    v8f o[4] = {};

    // Prime both pipeline stages (key blocks 0 and 32).
    v16bf kbuf[2][4], vbuf[2][4];
#pragma unroll
    for (int s = 0; s < 2; ++s) {
        const int kb = s * 32;
#pragma unroll
        for (int nt = 0; nt < 2; ++nt) {
            kbuf[s][nt * 2 + 0] = load_bfrag(k + (kb + nt * 16) * HDIM + 0,  HDIM, lane);
            kbuf[s][nt * 2 + 1] = load_bfrag(k + (kb + nt * 16) * HDIM + 32, HDIM, lane);
        }
#pragma unroll
        for (int t = 0; t < 4; ++t)
            vbuf[s][t] = load_bfrag(vt + (t * 16) * A_DIM + kb, A_DIM, lane);
    }

    auto step = [&](int kb, v16bf (&kc)[4], v16bf (&vc)[4]) {
        // S = Q @ K^T : 16 queries x 32 keys (two 16x16 C tiles, K=64 contraction)
        v8f s0 = {}, s1 = {};
        s0 = wmma_bf16(qa0, kc[0], s0);
        s0 = wmma_bf16(qa1, kc[1], s0);
        s1 = wmma_bf16(qa0, kc[2], s1);
        s1 = wmma_bf16(qa1, kc[3], s1);

        // Reload this K buffer for kb+64; must stay here (overlaps softmax).
        if (kb + 64 < A_DIM) {
#pragma unroll
            for (int nt = 0; nt < 2; ++nt) {
                kc[nt * 2 + 0] = load_bfrag(k + (kb + 64 + nt * 16) * HDIM + 0,  HDIM, lane);
                kc[nt * 2 + 1] = load_bfrag(k + (kb + 64 + nt * 16) * HDIM + 32, HDIM, lane);
            }
        }
        SCHED_BARRIER();

        // scale + relation bias + online softmax (row reductions in 16-lane groups)
        float p0[8], p1[8], alpha[8];
#pragma unroll
        for (int r = 0; r < 8; ++r) {
            const int qrow = qbase + r + 8 * half;
            float v0 = s0[r] * 0.125f + bias[qrow * A_DIM + kb + nl];
            float v1 = s1[r] * 0.125f + bias[qrow * A_DIM + kb + 16 + nl];
            float tm = fmaxf(v0, v1);
#pragma unroll
            for (int mk = 1; mk < 16; mk <<= 1) tm = fmaxf(tm, __shfl_xor(tm, mk, 32));
            const float nm = fmaxf(run_m[r], tm);
            alpha[r] = __expf(run_m[r] - nm);
            run_m[r] = nm;
            p0[r] = __expf(v0 - nm);
            p1[r] = __expf(v1 - nm);
            float rs = p0[r] + p1[r];
#pragma unroll
            for (int mk = 1; mk < 16; mk <<= 1) rs += __shfl_xor(rs, mk, 32);
            run_l[r] = run_l[r] * alpha[r] + rs;
        }
#pragma unroll
        for (int t = 0; t < 4; ++t)
#pragma unroll
            for (int r = 0; r < 8; ++r) o[t][r] *= alpha[r];

        // C-layout P -> LDS (row-major [m][key32]) -> A-layout fragment.
        __bf16* lp = &lds_p[wave][0];
#pragma unroll
        for (int r = 0; r < 8; ++r) {
            const int m = r + 8 * half;
            lp[m * 32 + nl]      = (__bf16)p0[r];
            lp[m * 32 + 16 + nl] = (__bf16)p1[r];
        }
        asm volatile("s_wait_dscnt 0" ::: "memory");
        __builtin_amdgcn_wave_barrier();

        const v16bf pa = load_afrag(lp, 32, lane);
#pragma unroll
        for (int t = 0; t < 4; ++t)
            o[t] = wmma_bf16(pa, vc[t], o[t]);   // O[16 x 64] += P @ V

        // Reload this V buffer for kb+64.
        if (kb + 64 < A_DIM) {
#pragma unroll
            for (int t = 0; t < 4; ++t)
                vc[t] = load_bfrag(vt + (t * 16) * A_DIM + kb + 64, A_DIM, lane);
        }
        SCHED_BARRIER();
        __builtin_amdgcn_wave_barrier();
    };

    for (int kb = 0; kb < A_DIM; kb += 64) {
        step(kb,      kbuf[0], vbuf[0]);
        step(kb + 32, kbuf[1], vbuf[1]);
    }

    // normalize and write [a][b][h*64+d] bf16
#pragma unroll
    for (int r = 0; r < 8; ++r) {
        const float inv = 1.0f / run_l[r];
        const int a_ = qbase + r + 8 * half;
#pragma unroll
        for (int t = 0; t < 4; ++t) {
            const int d = h * HDIM + t * 16 + nl;
            out[(a_ * B_DIM + b_) * DMODEL + d] = (__bf16)(o[t][r] * inv);
        }
    }
}

// ---------------------------------------------------------------------------
// Launch: cvt x5  ->  proj q/k/v  ->  flash attention  ->  proj out
// Workspace (bytes):
//   [0,16M)    src bf16               (reused as attn output bf16 afterwards)
//   [16M,18M)  Wq,Wk,Wv,Wo bf16 (512KB each)
//   [18M,34M)  q bf16   [b][h][a][d]
//   [34M,50M)  k bf16   [b][h][a][d]
//   [50M,66M)  v bf16   [b][h][d][a]
// ---------------------------------------------------------------------------
extern "C" void kernel_launch(void* const* d_in, const int* in_sizes, int n_in,
                              void* d_out, int out_size, void* d_ws, size_t ws_size,
                              hipStream_t stream) {
    const float* src  = (const float*)d_in[0];
    const float* rbias= (const float*)d_in[1];
    const float* Wq   = (const float*)d_in[2];
    const float* bq   = (const float*)d_in[3];
    const float* Wk   = (const float*)d_in[4];
    const float* bk   = (const float*)d_in[5];
    const float* Wv   = (const float*)d_in[6];
    const float* bv   = (const float*)d_in[7];
    const float* Wo   = (const float*)d_in[8];
    const float* bo   = (const float*)d_in[9];
    float* out = (float*)d_out;

    char* ws = (char*)d_ws;
    __bf16* srcbf = (__bf16*)(ws);
    __bf16* wqbf  = (__bf16*)(ws + (size_t)(16u << 20));
    __bf16* wkbf  = wqbf + DMODEL * DMODEL;
    __bf16* wvbf  = wkbf + DMODEL * DMODEL;
    __bf16* wobf  = wvbf + DMODEL * DMODEL;
    __bf16* qbf   = (__bf16*)(ws + (size_t)(18u << 20));
    __bf16* kbf   = (__bf16*)(ws + (size_t)(34u << 20));
    __bf16* vtbf  = (__bf16*)(ws + (size_t)(50u << 20));
    __bf16* attnbf = srcbf;  // src region is dead after the 3 projections

    const int nsrc = MROWS * DMODEL;     // 8388608
    const int nw   = DMODEL * DMODEL;    // 262144

    cvt_kernel<<<nsrc / 1024, 256, 0, stream>>>(src, srcbf, nsrc);
    cvt_kernel<<<nw / 1024,   256, 0, stream>>>(Wq, wqbf, nw);
    cvt_kernel<<<nw / 1024,   256, 0, stream>>>(Wk, wkbf, nw);
    cvt_kernel<<<nw / 1024,   256, 0, stream>>>(Wv, wvbf, nw);
    cvt_kernel<<<nw / 1024,   256, 0, stream>>>(Wo, wobf, nw);

    proj_kernel<<<1024, 256, 0, stream>>>(srcbf, wqbf, bq, qbf,  nullptr, 0);
    proj_kernel<<<1024, 256, 0, stream>>>(srcbf, wkbf, bk, kbf,  nullptr, 1);
    proj_kernel<<<1024, 256, 0, stream>>>(srcbf, wvbf, bv, vtbf, nullptr, 2);

    attn_kernel<<<128 * 16, 128, 0, stream>>>(qbf, kbf, vtbf, rbias, attnbf);

    proj_kernel<<<1024, 256, 0, stream>>>(attnbf, wobf, bo, nullptr, out, 3);
}
